// GATConvBinClass_52501680227001
// MI455X (gfx1250) — compile-verified, matrix operand order checked by reading, not run on hardware
//
#include <hip/hip_runtime.h>
#include <hip/hip_bf16.h>

typedef __attribute__((ext_vector_type(16))) __bf16 v16bf;
typedef __attribute__((ext_vector_type(8)))  __bf16 v8bf;
typedef __attribute__((ext_vector_type(8)))  float  v8f;

#define DH   256
#define GDIM 512
#define NEG_SLOPE 0.2f

// ---------------- ordered-float encoding for atomic max ----------------
__device__ __forceinline__ unsigned ord_enc(float f) {
    unsigned u = __float_as_uint(f);
    return (u & 0x80000000u) ? ~u : (u | 0x80000000u);
}
__device__ __forceinline__ float ord_dec(unsigned e) {
    return (e & 0x80000000u) ? __uint_as_float(e & 0x7fffffffu)
                             : __uint_as_float(~e);
}

// ---------------- fp32 -> bf16 converters ----------------
// contiguous convert (layer-1 activations)
__global__ void conv_x_bf16(const float* __restrict__ X, __bf16* __restrict__ Xb, int total) {
    int i = blockIdx.x * blockDim.x + threadIdx.x;
    if (i >= total) return;
    Xb[i] = (__bf16)X[i];
}
// transpose + convert weights: W[K,256] fp32 -> Wt[256,K] bf16
__global__ void conv_w_bf16(const float* __restrict__ W, __bf16* __restrict__ Wt, int K) {
    int i = blockIdx.x * blockDim.x + threadIdx.x;
    if (i >= K * DH) return;
    int c = i & (DH - 1);
    int k = i >> 8;
    Wt[(size_t)c * K + k] = (__bf16)W[i];
}

// ---------------- WMMA GEMM: H[N,256] = X[N,K] @ W[K,256] ----------------
// A = Xb (bf16 row-major [N,K]), B = Wt (bf16, W transposed: [256,K]).
// grid = (N/16, 4), block = 32 (one wave). Each wave computes a 16x64 C tile
// (4 accumulators) reusing one A fragment per K-step of 32.
// 16-bit fragment layout (wave32): half-wave h in {0,1} covers K = 8h..8h+7
// and 8h+16..8h+23 -> two contiguous 16B runs => two b128 loads per fragment.
__global__ __launch_bounds__(32)
void gat_gemm_wmma(const __bf16* __restrict__ Xb, const __bf16* __restrict__ Wt,
                   float* __restrict__ H, int Nrows, int K) {
    const int mBase = blockIdx.x * 16;
    const int nBase = blockIdx.y * 64;
    const int lane  = threadIdx.x;          // 0..31 (wave32)
    const int lr    = lane & 15;
    const int kHalf = (lane < 16) ? 0 : 8;

    int row = mBase + lr;
    if (row >= Nrows) row = Nrows - 1;      // N is a multiple of 16; defensive clamp

    v8f acc[4] = {{}, {}, {}, {}};

    for (int k0 = 0; k0 < K; k0 += 32) {
        const __bf16* abase = Xb + (size_t)row * K + k0 + kHalf;
        v8bf alo = *(const v8bf*)(abase);
        v8bf ahi = *(const v8bf*)(abase + 16);
        v16bf a = __builtin_shufflevector(alo, ahi,
                    0, 1, 2, 3, 4, 5, 6, 7, 8, 9, 10, 11, 12, 13, 14, 15);
#pragma unroll
        for (int t = 0; t < 4; ++t) {
            const int col = nBase + t * 16 + lr;
            const __bf16* bbase = Wt + (size_t)col * K + k0 + kHalf;
            v8bf blo = *(const v8bf*)(bbase);
            v8bf bhi = *(const v8bf*)(bbase + 16);
            v16bf b = __builtin_shufflevector(blo, bhi,
                        0, 1, 2, 3, 4, 5, 6, 7, 8, 9, 10, 11, 12, 13, 14, 15);
            acc[t] = __builtin_amdgcn_wmma_f32_16x16x32_bf16(
                false, a, false, b, (short)0, acc[t], false, false);
        }
    }

    // store C: lane 0-15 -> M = vgpr, lane 16-31 -> M = vgpr+8; N = lane&15
    const int mOff = (lane < 16) ? 0 : 8;
#pragma unroll
    for (int t = 0; t < 4; ++t) {
#pragma unroll
        for (int v = 0; v < 8; ++v) {
            int m = mBase + mOff + v;
            if (m < Nrows)
                H[(size_t)m * DH + nBase + t * 16 + lr] = acc[t][v];
        }
    }
}

// ---------------- per-node attention logits ----------------
__global__ void gat_alpha(const float* __restrict__ H,
                          const float* __restrict__ av_src,
                          const float* __restrict__ av_dst,
                          float* __restrict__ asrc, float* __restrict__ adst, int N) {
    int n = blockIdx.x * blockDim.x + threadIdx.x;
    if (n >= N) return;
    const float* h = H + (size_t)n * DH;
    float s1 = 0.f, s2 = 0.f;
#pragma unroll 4
    for (int f = 0; f < DH; ++f) {
        float v = h[f];
        s1 += v * av_src[f];
        s2 += v * av_dst[f];
    }
    asrc[n] = s1;
    adst[n] = s2;
}

__device__ __forceinline__ void edge_sd(const int* __restrict__ ei, int E, int e,
                                        int& s, int& d) {
    if (e < E) { s = ei[e]; d = ei[E + e]; }
    else       { s = e - E; d = e - E; }      // self-loops appended
}

// pass 1: e = leaky_relu(asrc[s]+adst[d]); segment max over dst
__global__ void gat_edge_max(const int* __restrict__ ei, int E, int ET,
                             const float* __restrict__ asrc, const float* __restrict__ adst,
                             float* __restrict__ ee, unsigned* __restrict__ menc) {
    int e = blockIdx.x * blockDim.x + threadIdx.x;
    if (e >= ET) return;
    int s, d; edge_sd(ei, E, e, s, d);
    float v = asrc[s] + adst[d];
    v = (v >= 0.f) ? v : NEG_SLOPE * v;
    ee[e] = v;
    atomicMax(menc + d, ord_enc(v));
}

// pass 2: p = exp(e - m[dst]); segment sum over dst
__global__ void gat_edge_exp(const int* __restrict__ ei, int E, int ET,
                             const float* __restrict__ ee, const unsigned* __restrict__ menc,
                             float* __restrict__ p, float* __restrict__ ssum) {
    int e = blockIdx.x * blockDim.x + threadIdx.x;
    if (e >= ET) return;
    int s, d; edge_sd(ei, E, e, s, d);
    float pe = __expf(ee[e] - ord_dec(menc[d]));
    p[e] = pe;
    atomicAdd(ssum + d, pe);
}

// pass 3: out[dst] += (p/s[dst]) * H[src]  (one wave per edge, 8 feats/lane)
__global__ __launch_bounds__(256)
void gat_edge_agg(const int* __restrict__ ei, int E, int ET,
                  const float* __restrict__ p, const float* __restrict__ ssum,
                  const float* __restrict__ H, float* __restrict__ agg) {
    int e    = (blockIdx.x * blockDim.x + threadIdx.x) >> 5;
    int lane = threadIdx.x & 31;
    if (e >= ET) return;
    int s, d; edge_sd(ei, E, e, s, d);
    float coef = p[e] / ssum[d];
    const float* hs = H + (size_t)s * DH;
    float* od = agg + (size_t)d * DH;
#pragma unroll
    for (int i = 0; i < 8; ++i) {
        int f = lane + 32 * i;
        atomicAdd(od + f, coef * hs[f]);
    }
}

// bias + relu; also emit bf16 copy for the next layer's WMMA GEMM
__global__ void gat_finalize(float* __restrict__ agg, const float* __restrict__ b,
                             __bf16* __restrict__ Xb, int total) {
    int i = blockIdx.x * blockDim.x + threadIdx.x;
    if (i >= total) return;
    float v = agg[i] + b[i & (DH - 1)];
    v = v > 0.f ? v : 0.f;
    agg[i] = v;
    Xb[i]  = (__bf16)v;
}

// global mean pool: accumulate sums and counts (one wave per node)
__global__ __launch_bounds__(256)
void gat_pool(const float* __restrict__ h, const int* __restrict__ batch,
              float* __restrict__ sums, float* __restrict__ counts, int N) {
    int n    = (blockIdx.x * blockDim.x + threadIdx.x) >> 5;
    int lane = threadIdx.x & 31;
    if (n >= N) return;
    int g = batch[n];
    const float* hn = h + (size_t)n * DH;
    float* sg = sums + (size_t)g * DH;
#pragma unroll
    for (int i = 0; i < 8; ++i) {
        int f = lane + 32 * i;
        atomicAdd(sg + f, hn[f]);
    }
    if (lane == 0) atomicAdd(counts + g, 1.0f);
}

// FC (256->2) + softmax, one thread per graph
__global__ void gat_head(const float* __restrict__ sums, const float* __restrict__ counts,
                         const float* __restrict__ Wfc, const float* __restrict__ bfc,
                         float* __restrict__ out, int G) {
    int g = blockIdx.x * blockDim.x + threadIdx.x;
    if (g >= G) return;
    float inv = 1.0f / fmaxf(counts[g], 1.0f);
    float l0 = bfc[0], l1 = bfc[1];
    const float* sg = sums + (size_t)g * DH;
#pragma unroll 4
    for (int f = 0; f < DH; ++f) {
        float pv = sg[f] * inv;
        l0 += pv * Wfc[f * 2 + 0];
        l1 += pv * Wfc[f * 2 + 1];
    }
    float mx = fmaxf(l0, l1);
    float e0 = __expf(l0 - mx), e1 = __expf(l1 - mx);
    float s = e0 + e1;
    out[g * 2 + 0] = e0 / s;
    out[g * 2 + 1] = e1 / s;
}

// ---------------- host-side orchestration ----------------
static inline size_t align256(size_t x) { return (x + 255) & ~(size_t)255; }

extern "C" void kernel_launch(void* const* d_in, const int* in_sizes, int n_in,
                              void* d_out, int out_size, void* d_ws, size_t ws_size,
                              hipStream_t stream) {
    (void)out_size; (void)ws_size;
    if (n_in < 17) return;

    const float* x    = (const float*)d_in[0];
    const int*   ei   = (const int*)d_in[1];
    const int*   batch= (const int*)d_in[2];
    const float* W[3]    = {(const float*)d_in[3], (const float*)d_in[7],  (const float*)d_in[11]};
    const float* avs[3]  = {(const float*)d_in[4], (const float*)d_in[8],  (const float*)d_in[12]};
    const float* avd[3]  = {(const float*)d_in[5], (const float*)d_in[9],  (const float*)d_in[13]};
    const float* bias[3] = {(const float*)d_in[6], (const float*)d_in[10], (const float*)d_in[14]};
    const float* Wfc = (const float*)d_in[15];
    const float* bfc = (const float*)d_in[16];
    float* out = (float*)d_out;

    const int DIN = 128;
    const int N  = in_sizes[0] / DIN;     // 50000
    const int E  = in_sizes[1] / 2;       // 800000
    const int ET = E + N;                 // with self-loops

    // workspace carve-up
    char* ws = (char*)d_ws;
    size_t off = 0;
    float*    bufH  = (float*)(ws + off);    off += align256((size_t)N * DH * 4);
    float*    bufX  = (float*)(ws + off);    off += align256((size_t)N * DH * 4);
    __bf16*   bufXb = (__bf16*)(ws + off);   off += align256((size_t)N * DH * 2);
    __bf16*   bufWt = (__bf16*)(ws + off);   off += align256((size_t)DH * DH * 2);
    float*    asrc  = (float*)(ws + off);    off += align256((size_t)N * 4);
    float*    adst  = (float*)(ws + off);    off += align256((size_t)N * 4);
    unsigned* menc  = (unsigned*)(ws + off); off += align256((size_t)N * 4);
    float*    ssum  = (float*)(ws + off);    off += align256((size_t)N * 4);
    float*    ee    = (float*)(ws + off);    off += align256((size_t)ET * 4);
    float*    pbuf  = (float*)(ws + off);    off += align256((size_t)ET * 4);
    float*    sums  = (float*)(ws + off);    off += align256((size_t)GDIM * DH * 4);
    float*    cnts  = (float*)(ws + off);    off += align256((size_t)GDIM * 4);

    dim3 blkWave(32), blkFlat(256);
    dim3 gemmGrid((N + 15) / 16, 4);
    int nodeBlocks  = (N + 255) / 256;
    int edgeBlocks  = (ET + 255) / 256;
    int aggBlocks   = (ET + 7) / 8;        // 8 waves (edges) per 256-thread block
    int poolBlocks  = (N + 7) / 8;
    int elemBlocks  = ((size_t)N * DH + 255) / 256;

    // layer-1 input -> bf16
    conv_x_bf16<<<(N * DIN + 255) / 256, blkFlat, 0, stream>>>(x, bufXb, N * DIN);

    int K = DIN;
    for (int layer = 0; layer < 3; ++layer) {
        conv_w_bf16<<<(K * DH + 255) / 256, blkFlat, 0, stream>>>(W[layer], bufWt, K);
        gat_gemm_wmma<<<gemmGrid, blkWave, 0, stream>>>(bufXb, bufWt, bufH, N, K);
        gat_alpha<<<nodeBlocks, blkFlat, 0, stream>>>(bufH, avs[layer], avd[layer], asrc, adst, N);
        hipMemsetAsync(menc, 0, (size_t)N * 4, stream);
        hipMemsetAsync(ssum, 0, (size_t)N * 4, stream);
        hipMemsetAsync(bufX, 0, (size_t)N * DH * 4, stream);
        gat_edge_max<<<edgeBlocks, blkFlat, 0, stream>>>(ei, E, ET, asrc, adst, ee, menc);
        gat_edge_exp<<<edgeBlocks, blkFlat, 0, stream>>>(ei, E, ET, ee, menc, pbuf, ssum);
        gat_edge_agg<<<aggBlocks, blkFlat, 0, stream>>>(ei, E, ET, pbuf, ssum, bufH, bufX);
        // writes fp32 activations (bufX) and bf16 copy (bufXb) for next GEMM
        gat_finalize<<<elemBlocks, blkFlat, 0, stream>>>(bufX, bias[layer], bufXb, N * DH);
        K = DH;
    }

    hipMemsetAsync(sums, 0, (size_t)GDIM * DH * 4, stream);
    hipMemsetAsync(cnts, 0, (size_t)GDIM * 4, stream);
    gat_pool<<<poolBlocks, blkFlat, 0, stream>>>(bufX, batch, sums, cnts, N);
    gat_head<<<(GDIM + 63) / 64, 64, 0, stream>>>(sums, cnts, Wfc, bfc, out, GDIM);
}
